// ConditionedPairUpdate_31937376813598
// MI455X (gfx1250) — matrix operand order repaired and checked by reading, not compiled, not run on hardware
//
#include <hip/hip_runtime.h>
#include <hip/hip_bf16.h>
#include <math.h>

// ---------------- problem constants ----------------
#define NTOK 256
#define CZ   128
#define NHEAD 4
#define CHD  32
#define CS   384
#define CCND 96
#define NRBF 39
#define NN   (NTOK*NTOK)
#define INF_ 1e8f
#define EPS_ 1e-5f

typedef __attribute__((ext_vector_type(16))) __bf16 v16bf;
typedef __attribute__((ext_vector_type(8)))  float  v8f;
typedef __attribute__((ext_vector_type(4)))  int    v4i_;

// ---------------- gfx1250 async global->LDS staging ----------------
#if defined(__has_builtin)
#if __has_builtin(__builtin_amdgcn_global_load_async_to_lds_b128)
#define HAVE_ASYNC_LDS 1
#endif
#if __has_builtin(__builtin_amdgcn_s_wait_asynccnt)
#define HAVE_WAIT_ASYNC 1
#endif
#endif

static __device__ __forceinline__ void async_ld16(const void* g, void* l) {
#ifdef HAVE_ASYNC_LDS
  __builtin_amdgcn_global_load_async_to_lds_b128(
      (__attribute__((address_space(1))) v4i_*)(unsigned long long)(size_t)g,
      (__attribute__((address_space(3))) v4i_*)(unsigned int)(size_t)l, 0, 0);
#else
  *(uint4*)l = *(const uint4*)g;
#endif
}
static __device__ __forceinline__ void wait_async() {
#ifdef HAVE_ASYNC_LDS
#ifdef HAVE_WAIT_ASYNC
  __builtin_amdgcn_s_wait_asynccnt(0);
#else
  __asm__ __volatile__("s_wait_asynccnt 0x0" ::: "memory");
#endif
#endif
}

// ---------------- small device helpers ----------------
static __device__ __forceinline__ float wsum(float v) {
#pragma unroll
  for (int o = 16; o; o >>= 1) v += __shfl_xor(v, o, 32);
  return v;
}
static __device__ __forceinline__ float wmaxf(float v) {
#pragma unroll
  for (int o = 16; o; o >>= 1) v = fmaxf(v, __shfl_xor(v, o, 32));
  return v;
}
static __device__ __forceinline__ float sigm(float x) { return 1.0f / (1.0f + __expf(-x)); }

// A-fragment (16x32 bf16, row-major source): lane%16 = M row;
// k = e + 16*(lane>=16) -> one contiguous 16-element run per lane
static __device__ __forceinline__ v16bf ldA(const __bf16* p, int stride, int row0, int k0, int lane) {
  v16bf a;
  int m  = row0 + (lane & 15);
  int kb = k0 + ((lane >> 4) << 4);
#pragma unroll
  for (int e = 0; e < 16; ++e)
    a[e] = p[m * stride + kb + e];
  return a;
}
// B-fragment (32x16) from a COLUMN-major source (src[n][k]): lane%16 = N col;
// k = e + 16*(lane>=16) -> one contiguous 16-element run per lane
static __device__ __forceinline__ v16bf ldBT(const __bf16* p, int stride, int k0, int col0, int lane) {
  v16bf b;
  int n  = col0 + (lane & 15);
  int kb = k0 + ((lane >> 4) << 4);
#pragma unroll
  for (int e = 0; e < 16; ++e)
    b[e] = p[n * stride + kb + e];
  return b;
}

// ---------------- weight cast / concat ----------------
__global__ void cast_cat_kernel(const float* s0, const float* s1, const float* s2, const float* s3,
                                __bf16* d, int K, int C, int nsrc) {
  int total = K * C * nsrc;
  for (int idx = blockIdx.x * blockDim.x + threadIdx.x; idx < total; idx += gridDim.x * blockDim.x) {
    int k = idx / (C * nsrc);
    int r = idx - k * (C * nsrc);
    int g = r / C;
    int c = r - g * C;
    const float* s = (g == 0) ? s0 : (g == 1) ? s1 : (g == 2) ? s2 : s3;
    d[idx] = (__bf16)s[k * C + c];
  }
}

// ---------------- edge bias: RBF path + projected-LN(edge) path ----------------
__global__ __launch_bounds__(256) void edge_bias_kernel(
    const float* coords, const float* edge,
    const float* rbf_w1, const float* rbf_g, const float* rbf_b, const float* rbf_w2,
    const float* th_g, const float* th_b, const float* th_w, float* eb) {
  int i = blockIdx.x, j = threadIdx.x;
  float dx = coords[i * 3 + 0] - coords[j * 3 + 0];
  float dy = coords[i * 3 + 1] - coords[j * 3 + 1];
  float dz = coords[i * 3 + 2] - coords[j * 3 + 2];
  float D = sqrtf(fmaxf(dx * dx + dy * dy + dz * dz, 1e-12f));
  const float dmin = 0.325f, dmax = 5.075f;
  const float step = (dmax - dmin) / (NRBF - 1);
  const float sigma = (dmax - dmin) / NRBF;
  float r[NRBF];
#pragma unroll
  for (int t = 0; t < NRBF; ++t) {
    float u = (D - (dmin + step * t)) / sigma;
    r[t] = __expf(-u * u);
  }
  float tv[32];
  float m = 0.f;
#pragma unroll
  for (int c = 0; c < 32; ++c) {
    float a = 0.f;
#pragma unroll
    for (int t = 0; t < NRBF; ++t) a += r[t] * rbf_w1[t * 32 + c];
    tv[c] = a;
    m += a;
  }
  m *= (1.0f / 32.0f);
  float v = 0.f;
#pragma unroll
  for (int c = 0; c < 32; ++c) { float d0 = tv[c] - m; v += d0 * d0; }
  v *= (1.0f / 32.0f);
  float inv = rsqrtf(v + EPS_);
  float db[4] = {0.f, 0.f, 0.f, 0.f};
#pragma unroll
  for (int c = 0; c < 32; ++c) {
    float xn = (tv[c] - m) * inv * rbf_g[c] + rbf_b[c];
#pragma unroll
    for (int h = 0; h < 4; ++h) db[h] += xn * rbf_w2[c * 4 + h];
  }
  const float* er = &edge[(size_t)(i * NTOK + j) * CZ];
  float s1 = 0.f, s2 = 0.f;
  for (int c = 0; c < CZ; ++c) { float x = er[c]; s1 += x; s2 += x * x; }
  float mm = s1 / CZ;
  float vv = s2 / CZ - mm * mm;
  float in2 = rsqrtf(vv + EPS_);
  float th[4] = {0.f, 0.f, 0.f, 0.f};
  for (int c = 0; c < CZ; ++c) {
    float xn = (er[c] - mm) * in2 * th_g[c] + th_b[c];
#pragma unroll
    for (int h = 0; h < 4; ++h) th[h] += xn * th_w[c * 4 + h];
  }
#pragma unroll
  for (int h = 0; h < 4; ++h) eb[((h * NTOK) + i) * NTOK + j] = db[h] + th[h];
}

// ---------------- LN(z) -> bf16 (optionally reading transposed z) ----------------
__global__ __launch_bounds__(256) void ln_z_kernel(const float* z, const float* g, const float* b,
                                                   __bf16* zl, int transpose) {
  int w = threadIdx.x >> 5, lane = threadIdx.x & 31;
  int vec = blockIdx.x * 8 + w;
  int i = vec >> 8, j = vec & 255;
  int src = transpose ? (j * NTOK + i) : vec;
  const float* zr = &z[(size_t)src * CZ];
  float s1 = 0.f, s2 = 0.f;
#pragma unroll
  for (int t = 0; t < 4; ++t) { float x = zr[lane + 32 * t]; s1 += x; s2 += x * x; }
  s1 = wsum(s1); s2 = wsum(s2);
  float m = s1 / CZ, v = s2 / CZ - m * m, inv = rsqrtf(v + EPS_);
#pragma unroll
  for (int t = 0; t < 4; ++t) {
    int c = lane + 32 * t;
    zl[(size_t)vec * CZ + c] = (__bf16)((zr[c] - m) * inv * g[c] + b[c]);
  }
}

// ---------------- generic WMMA GEMM (M=65536, block tile 32x64, 8 waves) ----------------
struct GemmArgs {
  const __bf16* A; int lda; int K;
  const __bf16* B; int ldb; int colOff;
  const __bf16* aux1; const __bf16* aux2;
  const float* bias;
  __bf16* o0; __bf16* o1; __bf16* o2; __bf16* o3;
  float* of; int ldo; int transpose;
};

template <int MODE>
__global__ __launch_bounds__(256) void gemm_kernel(GemmArgs ga) {
  __shared__ __bf16 As[32][256];   // A tile, row-major
  __shared__ __bf16 Bt[64][256];   // B tile, TRANSPOSED (Bt[n][k]) for contiguous frags
  int tid = threadIdx.x;
  int m0 = blockIdx.x * 32;
  int n0 = blockIdx.y * 64;
  int K = ga.K;
  __builtin_prefetch(ga.B, 0, 1);
  // A tile: async global->LDS, 16B per lane-op
  int kch = K >> 3;  // 8-element (16B) chunks per row
  for (int cid = tid; cid < 32 * kch; cid += 256) {
    int r = cid / kch, k8 = (cid - r * kch) << 3;
    async_ld16(&ga.A[(size_t)(m0 + r) * ga.lda + k8], &As[r][k8]);
  }
  // B tile: contiguous 16B global reads, transposed LDS stores
  for (int cid = tid; cid < K * 8; cid += 256) {
    int k = cid >> 3, c8 = (cid & 7) << 3;
    const __bf16* src = &ga.B[(size_t)k * ga.ldb + ga.colOff + n0 + c8];
#pragma unroll
    for (int e = 0; e < 8; ++e) Bt[c8 + e][k] = src[e];
  }
  wait_async();
  __syncthreads();
  int w = tid >> 5, lane = tid & 31;
  int tm = (w >> 2) * 16, tn = (w & 3) * 16;
  v8f acc = {};
  for (int kk = 0; kk < K; kk += 32) {
    v16bf a = ldA(&As[0][0], 256, tm, kk, lane);
    v16bf b = ldBT(&Bt[0][0], 256, kk, tn, lane);
    acc = __builtin_amdgcn_wmma_f32_16x16x32_bf16(false, a, false, b, (short)0, acc, false, false);
  }
#pragma unroll
  for (int r = 0; r < 8; ++r) {
    int ml = tm + r + ((lane >> 4) << 3);
    int nl = tn + (lane & 15);
    int m = m0 + ml, n = n0 + nl;
    float v = acc[r];
    if constexpr (MODE == 0) {            // qkvg split (+sigmoid on gate)
      if (n < 128)       ga.o0[(size_t)m * 128 + n] = (__bf16)v;
      else if (n < 256)  ga.o1[(size_t)m * 128 + n - 128] = (__bf16)v;
      else if (n < 384)  ga.o2[(size_t)m * 128 + n - 256] = (__bf16)v;
      else               ga.o3[(size_t)m * 128 + n - 384] = (__bf16)sigm(v);
    } else if constexpr (MODE == 1) {     // residual add (optional transpose back)
      int dr = ga.transpose ? ((m & 255) * NTOK + (m >> 8)) : m;
      ga.of[(size_t)dr * CZ + n] += v;
    } else if constexpr (MODE == 2) {     // sigmoid(x + bias) -> bf16
      ga.o0[(size_t)m * ga.ldo + n] = (__bf16)sigm(v + ga.bias[n]);
    } else if constexpr (MODE == 3) {     // a = scale*zn + shift
      ga.o0[(size_t)m * ga.ldo + n] =
          (__bf16)((float)ga.aux1[(size_t)m * ga.ldo + n] * (float)ga.aux2[(size_t)m * ga.ldo + n] + v);
    } else if constexpr (MODE == 5) {     // silu(x)
      ga.o0[(size_t)m * ga.ldo + n] = (__bf16)(v * sigm(v));
    } else if constexpr (MODE == 6) {     // aux1 * x
      ga.o0[(size_t)m * ga.ldo + n] = (__bf16)((float)ga.aux1[(size_t)m * ga.ldo + n] * v);
    } else if constexpr (MODE == 7) {     // z += gate * x
      ga.of[(size_t)m * CZ + n] += (float)ga.aux1[(size_t)m * CZ + n] * v;
    }
  }
}

// ---------------- fused attention: one block per (row i, head h) ----------------
__global__ __launch_bounds__(256) void attn_kernel(
    const __bf16* qb, const __bf16* kb, const __bf16* vb, const __bf16* gb,
    const float* eb, const float* mask, __bf16* og, int transpose) {
  extern __shared__ char smem[];
  __bf16* qs = (__bf16*)smem;                         // 256x32 bf16 row-major
  __bf16* ks = qs + NTOK * CHD;                       // 256x32 bf16 row-major
  __bf16* vt = ks + NTOK * CHD;                       // 32x256 bf16 (transposed V)
  float* S = (float*)(smem + 3 * NTOK * CHD * 2);     // 256x256 f32 scores
  float* rsum = (float*)(smem + 3 * NTOK * CHD * 2 + NTOK * NTOK * 4);
  int i = blockIdx.x >> 2, h = blockIdx.x & 3;
  int tid = threadIdx.x;
  __builtin_prefetch(gb + (size_t)(i * NTOK) * CZ + h * CHD, 0, 1);
  // q,k: async global->LDS (rows are 64B contiguous); v: transposed manual staging
  for (int cid = tid; cid < 1024; cid += 256) {       // 1024 x 16B chunks
    int j = cid >> 2, c8 = (cid & 3) << 3;
    size_t go = (size_t)(i * NTOK + j) * CZ + h * CHD + c8;
    async_ld16(&qb[go], &qs[j * CHD + c8]);
    async_ld16(&kb[go], &ks[j * CHD + c8]);
  }
  for (int idx = tid; idx < NTOK * CHD; idx += 256) {
    int j = idx >> 5, c = idx & 31;
    vt[c * NTOK + j] = vb[(size_t)(i * NTOK + j) * CZ + h * CHD + c];
  }
  wait_async();
  __syncthreads();
  int w = tid >> 5, lane = tid & 31;
  const float scale = 0.17677669529663687f;  // 1/sqrt(32)
  // S = q k^T * scale + edge_bias + mask_bias
  for (int t = w; t < 256; t += 8) {
    int tm = (t >> 4) * 16, tn = (t & 15) * 16;
    v16bf a = ldA(qs, CHD, tm, 0, lane);
    v16bf b = ldBT(ks, CHD, 0, tn, lane);   // B[k][n] = ks[n][k], contiguous in k
    v8f acc = {};
    acc = __builtin_amdgcn_wmma_f32_16x16x32_bf16(false, a, false, b, (short)0, acc, false, false);
#pragma unroll
    for (int r = 0; r < 8; ++r) {
      int qr = tm + r + ((lane >> 4) << 3);
      int kc = tn + (lane & 15);
      float ebv = transpose ? eb[(h * NTOK + kc) * NTOK + qr] : eb[(h * NTOK + qr) * NTOK + kc];
      float mv = transpose ? (mask[kc * NTOK + i] - 1.f) * INF_ : (mask[i * NTOK + kc] - 1.f) * INF_;
      S[qr * NTOK + kc] = acc[r] * scale + ebv + mv;
    }
  }
  __syncthreads();
  // row softmax (unnormalized exp into P, row sums kept for O epilogue)
  __bf16* P = (__bf16*)S;  // P row r lives in the first half of S row r
  for (int rr = 0; rr < 32; ++rr) {
    int r = w * 32 + rr;
    float mx = -1e30f;
#pragma unroll
    for (int t = 0; t < 8; ++t) mx = fmaxf(mx, S[r * NTOK + lane + 32 * t]);
    mx = wmaxf(mx);
    float ev[8];
    float sum = 0.f;
#pragma unroll
    for (int t = 0; t < 8; ++t) { ev[t] = __expf(S[r * NTOK + lane + 32 * t] - mx); sum += ev[t]; }
    sum = wsum(sum);
    if (lane == 0) rsum[r] = sum;
    __asm__ __volatile__("" ::: "memory");  // keep S reads before in-place P writes
#pragma unroll
    for (int t = 0; t < 8; ++t) P[r * 512 + lane + 32 * t] = (__bf16)ev[t];
  }
  __syncthreads();
  // O = P @ V, then gate & normalize
  for (int t = w; t < 32; t += 8) {
    int tm = (t >> 1) * 16, tn = (t & 1) * 16;
    v8f acc = {};
    for (int kk = 0; kk < NTOK; kk += 32) {
      v16bf a = ldA(P, 512, tm, kk, lane);
      v16bf b = ldBT(vt, NTOK, kk, tn, lane);  // B[k][n] = vt[n][k], contiguous in k
      acc = __builtin_amdgcn_wmma_f32_16x16x32_bf16(false, a, false, b, (short)0, acc, false, false);
    }
#pragma unroll
    for (int r = 0; r < 8; ++r) {
      int m = tm + r + ((lane >> 4) << 3);
      int n = tn + (lane & 15);
      float o = acc[r] / rsum[m];
      size_t go = (size_t)(i * NTOK + m) * CZ + h * CHD + n;
      og[go] = (__bf16)(o * (float)gb[go]);
    }
  }
}

// ---------------- cond = LN(node) @ cond_w ----------------
__global__ __launch_bounds__(128) void cond_kernel(const float* node, const float* g, const float* b,
                                                   const float* w, float* cond) {
  __shared__ float r1[128], r2[128];
  int n = blockIdx.x, tid = threadIdx.x;
  const float* nr = &node[(size_t)n * CS];
  float s1 = 0.f, s2 = 0.f;
  for (int t = tid; t < CS; t += 128) { float x = nr[t]; s1 += x; s2 += x * x; }
  r1[tid] = s1; r2[tid] = s2;
  __syncthreads();
  for (int s = 64; s; s >>= 1) {
    if (tid < s) { r1[tid] += r1[tid + s]; r2[tid] += r2[tid + s]; }
    __syncthreads();
  }
  float m = r1[0] / CS, v = r2[0] / CS - m * m, inv = rsqrtf(v + EPS_);
  if (tid < CCND) {
    float acc = 0.f;
    for (int c = 0; c < CS; ++c) {
      float xn = (nr[c] - m) * inv * g[c] + b[c];
      acc += xn * w[c * CCND + tid];
    }
    cond[n * CCND + tid] = acc;
  }
}

// ---------------- sl = LN(pair_cond), zn = LN_noaffine(z) ----------------
__global__ __launch_bounds__(256) void sl_zn_kernel(const float* z, const float* cond,
                                                    const float* tg, const float* tb,
                                                    __bf16* sl, __bf16* zn) {
  int w = threadIdx.x >> 5, lane = threadIdx.x & 31;
  int vec = blockIdx.x * 8 + w;
  int i = vec >> 8, j = vec & 255;
  float pv[6];
  float s1 = 0.f, s2 = 0.f;
#pragma unroll
  for (int t = 0; t < 6; ++t) {
    int c = lane + 32 * t;
    float x = (c < CCND) ? cond[i * CCND + c] : cond[j * CCND + c - CCND];
    pv[t] = x; s1 += x; s2 += x * x;
  }
  s1 = wsum(s1); s2 = wsum(s2);
  float m = s1 / 192.f, v = s2 / 192.f - m * m, inv = rsqrtf(v + EPS_);
#pragma unroll
  for (int t = 0; t < 6; ++t) {
    int c = lane + 32 * t;
    sl[(size_t)vec * 192 + c] = (__bf16)((pv[t] - m) * inv * tg[c] + tb[c]);
  }
  const float* zr = &z[(size_t)vec * CZ];
  float z1 = 0.f, z2 = 0.f;
#pragma unroll
  for (int t = 0; t < 4; ++t) { float x = zr[lane + 32 * t]; z1 += x; z2 += x * x; }
  z1 = wsum(z1); z2 = wsum(z2);
  float zm = z1 / CZ, zv = z2 / CZ - zm * zm, zi = rsqrtf(zv + EPS_);
#pragma unroll
  for (int t = 0; t < 4; ++t) {
    int c = lane + 32 * t;
    zn[(size_t)vec * CZ + c] = (__bf16)((zr[c] - zm) * zi);
  }
}

// ---------------- host orchestration ----------------
extern "C" void kernel_launch(void* const* d_in, const int* in_sizes, int n_in,
                              void* d_out, int out_size, void* d_ws, size_t ws_size,
                              hipStream_t stream) {
  (void)in_sizes; (void)n_in; (void)out_size; (void)ws_size;
  const float* node   = (const float*)d_in[0];
  const float* edge   = (const float*)d_in[1];
  const float* coords = (const float*)d_in[2];
  const float* emask  = (const float*)d_in[3];
#define PRM(k) ((const float*)d_in[4 + (k)])
  float* zout = (float*)d_out;
  char* ws = (char*)d_ws;
  const size_t MB = 1ull << 20;
  float*  eb   = (float*)(ws);                 // 1 MB : [4][256][256]
  __bf16* wreg = (__bf16*)(ws + 1 * MB);       // bf16 weights (~0.7 MB)
  float*  cond = (float*)(ws + 2 * MB);        // [256][96]
  char* big = ws + 4 * MB;
  __bf16* zl = (__bf16*)(big);
  __bf16* qb = (__bf16*)(big + 16 * MB);
  __bf16* kb = (__bf16*)(big + 32 * MB);
  __bf16* vb = (__bf16*)(big + 48 * MB);
  __bf16* gb = (__bf16*)(big + 64 * MB);
  __bf16* og = (__bf16*)(big + 80 * MB);
  __bf16* slb = (__bf16*)(big);
  __bf16* znb = (__bf16*)(big + 24 * MB);
  __bf16* ab  = (__bf16*)(big + 40 * MB);
  __bf16* gtb = (__bf16*)(big + 56 * MB);
  __bf16* bh  = (__bf16*)(big + 72 * MB);
  __bf16* Wqkvg_s = wreg;
  __bf16* Wo_s    = wreg + 65536;
  __bf16* Wqkvg_e = wreg + 81920;
  __bf16* Wo_e    = wreg + 147456;
  __bf16* Wssg    = wreg + 163840;   // [192][384] = scale|shift|gate
  __bf16* W12     = wreg + 237568;   // [128][512] = w1|w2
  __bf16* Wto     = wreg + 303104;   // [256][128]

  (void)hipMemcpyAsync(zout, edge, (size_t)NN * CZ * sizeof(float), hipMemcpyDeviceToDevice, stream);

  cast_cat_kernel<<<256, 256, 0, stream>>>(PRM(22), PRM(23), PRM(24), PRM(25), Wqkvg_s, 128, 128, 4);
  cast_cat_kernel<<<64, 256, 0, stream>>>(PRM(26), nullptr, nullptr, nullptr, Wo_s, 128, 128, 1);
  cast_cat_kernel<<<256, 256, 0, stream>>>(PRM(29), PRM(30), PRM(31), PRM(32), Wqkvg_e, 128, 128, 4);
  cast_cat_kernel<<<64, 256, 0, stream>>>(PRM(33), nullptr, nullptr, nullptr, Wo_e, 128, 128, 1);
  cast_cat_kernel<<<256, 256, 0, stream>>>(PRM(12), PRM(14), PRM(17), nullptr, Wssg, 192, 128, 3);
  cast_cat_kernel<<<256, 256, 0, stream>>>(PRM(15), PRM(16), nullptr, nullptr, W12, 128, 256, 2);
  cast_cat_kernel<<<128, 256, 0, stream>>>(PRM(19), nullptr, nullptr, nullptr, Wto, 256, 128, 1);

  edge_bias_kernel<<<256, 256, 0, stream>>>(coords, edge, PRM(0), PRM(1), PRM(2), PRM(3),
                                            PRM(4), PRM(5), PRM(6), eb);

  for (int pass = 0; pass < 2; ++pass) {
    const float* lng = pass ? PRM(27) : PRM(20);
    const float* lnb = pass ? PRM(28) : PRM(21);
    __bf16* Wqkvg = pass ? Wqkvg_e : Wqkvg_s;
    __bf16* Wo    = pass ? Wo_e : Wo_s;
    ln_z_kernel<<<8192, 256, 0, stream>>>(zout, lng, lnb, zl, pass);
    GemmArgs g0 = {zl, 128, 128, Wqkvg, 512, 0, nullptr, nullptr, nullptr,
                   qb, kb, vb, gb, nullptr, 128, 0};
    gemm_kernel<0><<<dim3(2048, 8), 256, 0, stream>>>(g0);
    attn_kernel<<<1024, 256, 312320, stream>>>(qb, kb, vb, gb, eb, emask, og, pass);
    GemmArgs g1 = {og, 128, 128, Wo, 128, 0, nullptr, nullptr, nullptr,
                   nullptr, nullptr, nullptr, nullptr, zout, 128, pass};
    gemm_kernel<1><<<dim3(2048, 2), 256, 0, stream>>>(g1);
  }

  cond_kernel<<<256, 128, 0, stream>>>(node, PRM(7), PRM(8), PRM(9), cond);
  sl_zn_kernel<<<8192, 256, 0, stream>>>(zout, cond, PRM(10), PRM(11), slb, znb);
  GemmArgs gs = {slb, 192, 192, Wssg, 384, 0, nullptr, nullptr, PRM(13),
                 ab, nullptr, nullptr, nullptr, nullptr, 128, 0};
  gemm_kernel<2><<<dim3(2048, 2), 256, 0, stream>>>(gs);
  GemmArgs gsh = {slb, 192, 192, Wssg, 384, 128, ab, znb, nullptr,
                  ab, nullptr, nullptr, nullptr, nullptr, 128, 0};
  gemm_kernel<3><<<dim3(2048, 2), 256, 0, stream>>>(gsh);
  GemmArgs gg = {slb, 192, 192, Wssg, 384, 256, nullptr, nullptr, PRM(18),
                 gtb, nullptr, nullptr, nullptr, nullptr, 128, 0};
  gemm_kernel<2><<<dim3(2048, 2), 256, 0, stream>>>(gg);
  GemmArgs g1a = {ab, 128, 128, W12, 512, 0, nullptr, nullptr, nullptr,
                  bh, nullptr, nullptr, nullptr, nullptr, 256, 0};
  gemm_kernel<5><<<dim3(2048, 4), 256, 0, stream>>>(g1a);
  GemmArgs g2a = {ab, 128, 128, W12, 512, 256, bh, nullptr, nullptr,
                  bh, nullptr, nullptr, nullptr, nullptr, 256, 0};
  gemm_kernel<6><<<dim3(2048, 4), 256, 0, stream>>>(g2a);
  GemmArgs gto = {bh, 256, 256, Wto, 128, 0, gtb, nullptr, nullptr,
                  nullptr, nullptr, nullptr, nullptr, zout, 128, 0};
  gemm_kernel<7><<<dim3(2048, 2), 256, 0, stream>>>(gto);
#undef PRM
}
